// Cls_Classifier_17609365913787
// MI455X (gfx1250) — matrix operand order repaired and checked by reading, not compiled, not run on hardware
//
#include <hip/hip_runtime.h>
#include <hip/hip_bf16.h>
#include <stdint.h>

#define BATCH 4096
#define TSEQ  22
#define EMB   300
#define HID   300
#define KPAD  320      // K (=300) zero-padded to a multiple of 32 for 16x16x32 WMMA
#define G4H   1200     // 4*HID gate width (logical)
#define G4HP  1216     // gate width padded to 2x16-tile granularity for 2x2 blocking

typedef __attribute__((ext_vector_type(16))) _Float16 v16h;
typedef __attribute__((ext_vector_type(8)))  _Float16 v8h;
typedef __attribute__((ext_vector_type(8)))  float    v8f;

// ---------------------------------------------------------------------------
// Prep: f32 -> f16 weight conversion, K padded 300->320, N padded 1200->1216.
// W row n == column n of W^T, so the f16 copy is the WMMA B-matrix source.
// ---------------------------------------------------------------------------
__global__ void k_prep_weights(const float* __restrict__ Wih,
                               const float* __restrict__ Whh,
                               _Float16* __restrict__ WihH,
                               _Float16* __restrict__ WhhH) {
    int n = blockIdx.x;       // 0..1215
    int k = threadIdx.x;      // 0..319
    float a = (n < G4H && k < EMB) ? Wih[n * EMB + k] : 0.f;
    float b = (n < G4H && k < HID) ? Whh[n * HID + k] : 0.f;
    WihH[(size_t)n * KPAD + k] = (_Float16)a;
    WhhH[(size_t)n * KPAD + k] = (_Float16)b;
}

// ---------------------------------------------------------------------------
// Embedding gather -> f16 activation matrix X[t][b][KPAD] (zeros in pad).
// ---------------------------------------------------------------------------
__global__ void k_embed(const int* __restrict__ cap,
                        const float* __restrict__ embed,
                        _Float16* __restrict__ X) {
    int row = blockIdx.x;                 // 0..B*T-1, layout [T][B]
    int t = row / BATCH;
    int b = row - t * BATCH;
    int k = threadIdx.x;                  // 0..319
    int tok = cap[b * TSEQ + t];
    float v = (k < EMB) ? embed[(size_t)tok * EMB + k] : 0.f;
    X[(size_t)row * KPAD + k] = (_Float16)v;
}

// h0 = 0 (f16, incl. pad) and c0 = 0
__global__ void k_init(_Float16* __restrict__ Hh, float* __restrict__ C) {
    int i = blockIdx.x * blockDim.x + threadIdx.x;
    if (i < BATCH * KPAD) Hh[i] = (_Float16)0.f;
    if (i < BATCH * HID)  C[i]  = 0.f;
}

// ---------------------------------------------------------------------------
// Fragment loaders for 16-bit WMMA, contiguous k order k = 16*(lane/16) + s in both A and B.
// A (16x32, row-major [.,KPAD]): lane-half k-offset ak=0/16, 16 contiguous halves (two b128 loads).
// B (32x16 of W^T = 16 contiguous halves of W row): lane-half bk=0/16.
// ---------------------------------------------------------------------------
__device__ __forceinline__ v16h load_afrag(const _Float16* __restrict__ p) {
    v8h lo = *(const v8h*)(p);
    v8h hi = *(const v8h*)(p + 8);
    return __builtin_shufflevector(lo, hi,
           0,1,2,3,4,5,6,7,8,9,10,11,12,13,14,15);
}
__device__ __forceinline__ v16h load_bfrag(const _Float16* __restrict__ p) {
    return *(const v16h*)(p);
}

// One K-sweep of 2x2 register-blocked 16x16x32 WMMA: 4 MMAs per 4 fragment
// loads (2 A + 2 B) -> 2x the FLOP/byte of a 1x1 tile.
__device__ __forceinline__ void mma_sweep_2x2(const _Float16* __restrict__ A0,
                                              const _Float16* __restrict__ A1,
                                              const _Float16* __restrict__ B0,
                                              const _Float16* __restrict__ B1,
                                              int ak, int bk,
                                              v8f& c00, v8f& c01,
                                              v8f& c10, v8f& c11) {
#pragma unroll
    for (int kb = 0; kb < KPAD; kb += 32) {
        v16h a0 = load_afrag(A0 + kb + ak);
        v16h a1 = load_afrag(A1 + kb + ak);
        v16h b0 = load_bfrag(B0 + kb + bk);
        v16h b1 = load_bfrag(B1 + kb + bk);
        c00 = __builtin_amdgcn_wmma_f32_16x16x32_f16(false, a0, false, b0,
                                                     (short)0, c00, false, false);
        c01 = __builtin_amdgcn_wmma_f32_16x16x32_f16(false, a0, false, b1,
                                                     (short)0, c01, false, false);
        c10 = __builtin_amdgcn_wmma_f32_16x16x32_f16(false, a1, false, b0,
                                                     (short)0, c10, false, false);
        c11 = __builtin_amdgcn_wmma_f32_16x16x32_f16(false, a1, false, b1,
                                                     (short)0, c11, false, false);
    }
}

// ---------------------------------------------------------------------------
// gates[B][G4HP] = x_t @ W_ih^T + h @ W_hh^T   (biases added in cell kernel)
// Wave computes a 32x32 tile (2x2 of 16x16). Block = 8 waves as 4(M) x 2(N)
// -> 128x64 per block. Grid = (4096/128, 1216/64) = (32, 19).
// ---------------------------------------------------------------------------
__global__ __launch_bounds__(256)
void k_gates(const _Float16* __restrict__ X,    // [BATCH][KPAD] x_t slab
             const _Float16* __restrict__ Wih,  // [G4HP][KPAD]
             const _Float16* __restrict__ Hh,   // [BATCH][KPAD]
             const _Float16* __restrict__ Whh,  // [G4HP][KPAD]
             float* __restrict__ gates)         // [BATCH][G4HP]
{
    const int lane = threadIdx.x & 31;
    const int wave = threadIdx.x >> 5;
    const int wm = wave & 3, wn = wave >> 2;
    const int m0 = blockIdx.x * 128 + wm * 32;
    const int n0 = blockIdx.y * 64  + wn * 32;

    const int am = lane & 15;
    const int ak = (lane < 16) ? 0 : 16;
    const int bn = lane & 15;
    const int bk = (lane < 16) ? 0 : 16;

    const _Float16* A0x = X   + (size_t)(m0      + am) * KPAD;
    const _Float16* A1x = X   + (size_t)(m0 + 16 + am) * KPAD;
    const _Float16* A0h = Hh  + (size_t)(m0      + am) * KPAD;
    const _Float16* A1h = Hh  + (size_t)(m0 + 16 + am) * KPAD;
    const _Float16* B0i = Wih + (size_t)(n0      + bn) * KPAD;
    const _Float16* B1i = Wih + (size_t)(n0 + 16 + bn) * KPAD;
    const _Float16* B0h = Whh + (size_t)(n0      + bn) * KPAD;
    const _Float16* B1h = Whh + (size_t)(n0 + 16 + bn) * KPAD;

    v8f c00 = {}, c01 = {}, c10 = {}, c11 = {};
    mma_sweep_2x2(A0x, A1x, B0i, B1i, ak, bk, c00, c01, c10, c11);
    mma_sweep_2x2(A0h, A1h, B0h, B1h, ak, bk, c00, c01, c10, c11);

    // C/D layout: VGPR r, lanes 0-15 -> M=r, lanes 16-31 -> M=8+r; N = lane&15
    const int rb = (lane < 16) ? 0 : 8;
    const int nn = lane & 15;
#pragma unroll
    for (int r = 0; r < 8; ++r) {
        gates[(size_t)(m0      + rb + r) * G4HP + n0      + nn] = c00[r];
        gates[(size_t)(m0      + rb + r) * G4HP + n0 + 16 + nn] = c01[r];
        gates[(size_t)(m0 + 16 + rb + r) * G4HP + n0      + nn] = c10[r];
        gates[(size_t)(m0 + 16 + rb + r) * G4HP + n0 + 16 + nn] = c11[r];
    }
}

// ---------------------------------------------------------------------------
// Fused LSTM cell: nonlinearity, c/h update, f16 h for next step's GEMM,
// and scatter of h into `last` for rows whose sequence ends at step t.
// ---------------------------------------------------------------------------
__global__ __launch_bounds__(256)
void k_cell(const float* __restrict__ gates,   // [BATCH][G4HP]
            const float* __restrict__ b_ih,
            const float* __restrict__ b_hh,
            float* __restrict__ C,
            _Float16* __restrict__ Hh,
            const int* __restrict__ cap_len,
            float* __restrict__ last,
            int t)
{
    int idx = blockIdx.x * blockDim.x + threadIdx.x;
    if (idx >= BATCH * HID) return;
    int b = idx / HID;
    int j = idx - b * HID;
    const float* g = gates + (size_t)b * G4HP;
    float gi = g[j]         + b_ih[j]         + b_hh[j];
    float gf = g[HID + j]   + b_ih[HID + j]   + b_hh[HID + j];
    float gg = g[2*HID + j] + b_ih[2*HID + j] + b_hh[2*HID + j];
    float go = g[3*HID + j] + b_ih[3*HID + j] + b_hh[3*HID + j];
    float i_ = 1.f / (1.f + __expf(-gi));
    float f_ = 1.f / (1.f + __expf(-gf));
    float o_ = 1.f / (1.f + __expf(-go));
    float c_ = f_ * C[idx] + i_ * tanhf(gg);
    float h_ = o_ * tanhf(c_);
    C[idx] = c_;
    Hh[(size_t)b * KPAD + j] = (_Float16)h_;   // pad cols stay zero from k_init
    if (cap_len[b] - 1 == t) last[idx] = h_;
}

// ---------------------------------------------------------------------------
// Weight-norm head prep: Wcls[j][k] = g[j] * v[j][k] / ||v[j]||  (wave32 reduce)
// ---------------------------------------------------------------------------
__global__ void k_prep_head(const float* __restrict__ v_wn,
                            const float* __restrict__ g_wn,
                            float* __restrict__ Wcls) {
    int j    = threadIdx.x >> 5;   // 2 waves, one per class row
    int lane = threadIdx.x & 31;
    float s = 0.f;
    for (int k = lane; k < HID; k += 32) {
        float v = v_wn[j * HID + k];
        s += v * v;
    }
#pragma unroll
    for (int m = 16; m; m >>= 1) s += __shfl_xor(s, m, 32);
    float scale = g_wn[j] * rsqrtf(s);
    for (int k = lane; k < HID; k += 32)
        Wcls[j * HID + k] = scale * v_wn[j * HID + k];
}

// out[b][j] = last[b] . Wcls[j] + b_cls[j] ; one wave32 per batch row
__global__ __launch_bounds__(256)
void k_head(const float* __restrict__ last,
            const float* __restrict__ Wcls,
            const float* __restrict__ b_cls,
            float* __restrict__ out) {
    int wave = threadIdx.x >> 5;
    int lane = threadIdx.x & 31;
    int b = blockIdx.x * 8 + wave;
    const float* row = last + (size_t)b * HID;
    float s0 = 0.f, s1 = 0.f;
    for (int k = lane; k < HID; k += 32) {
        float x = row[k];
        s0 += x * Wcls[k];
        s1 += x * Wcls[HID + k];
    }
#pragma unroll
    for (int m = 16; m; m >>= 1) {
        s0 += __shfl_xor(s0, m, 32);
        s1 += __shfl_xor(s1, m, 32);
    }
    if (lane == 0) {
        out[b * 2 + 0] = s0 + b_cls[0];
        out[b * 2 + 1] = s1 + b_cls[1];
    }
}

// ---------------------------------------------------------------------------
extern "C" void kernel_launch(void* const* d_in, const int* in_sizes, int n_in,
                              void* d_out, int out_size, void* d_ws, size_t ws_size,
                              hipStream_t stream) {
    const int*   cap     = (const int*)  d_in[0];
    const int*   cap_len = (const int*)  d_in[1];
    const float* embed   = (const float*)d_in[2];
    const float* W_ih    = (const float*)d_in[3];
    const float* W_hh    = (const float*)d_in[4];
    const float* b_ih    = (const float*)d_in[5];
    const float* b_hh    = (const float*)d_in[6];
    const float* v_wn    = (const float*)d_in[7];
    const float* g_wn    = (const float*)d_in[8];
    const float* b_cls   = (const float*)d_in[9];
    float* out = (float*)d_out;

    char* ws = (char*)d_ws;
    size_t off = 0;
    auto carve = [&](size_t bytes) -> char* {
        char* p = ws + off;
        off = (off + bytes + 255) & ~(size_t)255;
        return p;
    };
    _Float16* WihH = (_Float16*)carve((size_t)G4HP * KPAD * sizeof(_Float16));
    _Float16* WhhH = (_Float16*)carve((size_t)G4HP * KPAD * sizeof(_Float16));
    _Float16* Xall = (_Float16*)carve((size_t)TSEQ * BATCH * KPAD * sizeof(_Float16));
    _Float16* Hh   = (_Float16*)carve((size_t)BATCH * KPAD * sizeof(_Float16));
    float*    C    = (float*)   carve((size_t)BATCH * HID * sizeof(float));
    float*    gts  = (float*)   carve((size_t)BATCH * G4HP * sizeof(float));
    float*    last = (float*)   carve((size_t)BATCH * HID * sizeof(float));
    float*    Wcls = (float*)   carve((size_t)2 * HID * sizeof(float));

    k_prep_weights<<<G4HP, KPAD, 0, stream>>>(W_ih, W_hh, WihH, WhhH);
    k_embed<<<BATCH * TSEQ, KPAD, 0, stream>>>(cap, embed, Xall);
    k_init<<<(BATCH * KPAD + 255) / 256, 256, 0, stream>>>(Hh, C);

    dim3 gGrid(BATCH / 128, G4HP / 64);   // (32, 19) blocks of 8 waves
    for (int t = 0; t < TSEQ; ++t) {
        k_gates<<<gGrid, 256, 0, stream>>>(Xall + (size_t)t * BATCH * KPAD,
                                           WihH, Hh, WhhH, gts);
        k_cell<<<(BATCH * HID + 255) / 256, 256, 0, stream>>>(
            gts, b_ih, b_hh, C, Hh, cap_len, last, t);
    }

    k_prep_head<<<1, 64, 0, stream>>>(v_wn, g_wn, Wcls);
    k_head<<<BATCH / 8, 256, 0, stream>>>(last, Wcls, b_cls, out);
}
